// DTFABlock_66675072303713
// MI455X (gfx1250) — matrix operand order repaired and checked by reading, not compiled, not run on hardware
//
#include <hip/hip_runtime.h>
#include <hip/hip_bf16.h>

#define B_   8
#define C_   64
#define F_   256
#define T_   512
#define FT_  131072          // F_*T_
#define NP_  1048576         // B_*FT_
#define NWG_D 1024
#define ITERS_D 8
#define ROWH 136             // padded f16 row (128 + 8) to avoid LDS bank conflicts

typedef _Float16 f16;
typedef __attribute__((ext_vector_type(16))) _Float16 v16h;
typedef __attribute__((ext_vector_type(8)))  _Float16 v8h;
typedef __attribute__((ext_vector_type(4)))  _Float16 v4h;
typedef __attribute__((ext_vector_type(8)))  float    v8f;
typedef __attribute__((ext_vector_type(4)))  float    v4f;

// ---------------- Kernel 1a: mean over T (rows) -> mT[b][c][f] ----------------
__global__ __launch_bounds__(256) void k_row_mean(const float* __restrict__ x,
                                                  float* __restrict__ mT) {
    int row  = blockIdx.x * 8 + (threadIdx.x >> 5);   // (b*C+c)*F + f
    int lane = threadIdx.x & 31;
    const float* p = x + (size_t)row * T_;
    v4f s4 = {0.f, 0.f, 0.f, 0.f};
#pragma unroll
    for (int q = 0; q < 4; ++q)
        s4 += *(const v4f*)(p + (lane + 32 * q) * 4);
    float s = s4[0] + s4[1] + s4[2] + s4[3];
#pragma unroll
    for (int m = 16; m >= 1; m >>= 1) s += __shfl_xor(s, m, 32);
    if (lane == 0) mT[row] = s * (1.0f / T_);
}

// ---------------- Kernel 1b: mean over F (cols) -> mF[b][c][t] ----------------
__global__ __launch_bounds__(256) void k_col_mean(const float* __restrict__ x,
                                                  float* __restrict__ mF) {
    int col = blockIdx.x * 256 + threadIdx.x;         // (b*C+c)*T + t
    int bc = col >> 9, t = col & 511;
    const float* p = x + (size_t)bc * FT_ + t;
    float s = 0.f;
    for (int f = 0; f < F_; ++f) s += p[f * T_];
    mF[col] = s * (1.0f / F_);
}

// ---------------- Kernel 2: tiny channel MLP -> zt[b][c][t], u[b][c][f]=wc*zf --
__global__ __launch_bounds__(256) void k_mlp(const float* __restrict__ mF,
                                             const float* __restrict__ mT,
                                             const float* __restrict__ w1,
                                             const float* __restrict__ b1,
                                             const float* __restrict__ w2,
                                             const float* __restrict__ b2,
                                             const float* __restrict__ wc,
                                             float* __restrict__ zt,
                                             float* __restrict__ u) {
    int p = blockIdx.x * 256 + threadIdx.x;
    const float* src; float* dst; int base, stride; bool isT;
    if (p < B_ * T_) {
        int b = p >> 9, t = p & 511;
        src = mF; dst = zt; base = b * C_ * T_ + t; stride = T_; isT = true;
    } else {
        int p2 = p - B_ * T_;
        if (p2 >= B_ * F_) return;
        int b = p2 >> 8, f = p2 & 255;
        src = mT; dst = u; base = b * C_ * F_ + f; stride = F_; isT = false;
    }
    float h[5];
#pragma unroll
    for (int o = 0; o < 5; ++o) h[o] = b1[o];
    for (int c = 0; c < C_; ++c) {
        float z = src[base + c * stride];
#pragma unroll
        for (int o = 0; o < 5; ++o) h[o] += w1[o * C_ + c] * z;
    }
#pragma unroll
    for (int o = 0; o < 5; ++o) h[o] = fmaxf(h[o], 0.f);
    for (int c = 0; c < C_; ++c) {
        float s = b2[c];
#pragma unroll
        for (int o = 0; o < 5; ++o) s += w2[c * 5 + o] * h[o];
        float sg = 1.f / (1.f + __expf(-s));
        dst[base + c * stride] = isT ? sg : wc[c] * sg;
    }
}

// ---------------- Kernel 3: att_map[b,f,t] = sum_c u[b,c,f]*zt[b,c,t] + bc -----
__global__ __launch_bounds__(256) void k_attmap(const float* __restrict__ u,
                                                const float* __restrict__ zt,
                                                const float* __restrict__ bc,
                                                float* __restrict__ att) {
    float bcv = bc[0];
    for (int idx = blockIdx.x * 256 + threadIdx.x; idx < NP_; idx += 262144) {
        int b = idx >> 17, rem = idx & (FT_ - 1);
        int f = rem >> 9, t = rem & 511;
        const float* up = u  + b * (C_ * F_) + f;
        const float* zp = zt + b * (C_ * T_) + t;
        float s = bcv;
#pragma unroll 8
        for (int c = 0; c < C_; ++c) s += up[c * F_] * zp[c * T_];
        att[idx] = s;
    }
}

// ---------------- Kernel 4: main WMMA GEMM + BN partial stats ------------------
// o[c,pix] = sum_k wo[c,k] * Bmat[k,pix] + bo[c], Bmat = [x ; (a*P[c,j]+bc2)*x]
__global__ __launch_bounds__(256) void k_dtfa_main(const float* __restrict__ x,
                                                   const float* __restrict__ att_map,
                                                   const float* __restrict__ wc2,
                                                   const float* __restrict__ bc2,
                                                   const float* __restrict__ wo,
                                                   const float* __restrict__ bo,
                                                   float* __restrict__ o_out,
                                                   float* __restrict__ partial) {
    __shared__ __align__(16) unsigned char smem[61952];
    f16*   woL    = (f16*)smem;                    // [64][ROWH] f16 (17408 B)
    f16*   bmat   = (f16*)(smem + 17408);          // [128][ROWH] f16 (34816 B)
    float* o_tile = (float*)(smem + 17408);        // alias: [64][128] f32
    float* P_s    = (float*)(smem + 52224);        // [64][24]
    float* bc2_s  = (float*)(smem + 58368);
    float* bo_s   = (float*)(smem + 58624);
    float* a_s    = (float*)(smem + 58880);        // [128]
    int*   j_s    = (int*)  (smem + 59392);        // [128]
    float* red    = (float*)(smem + 59904);        // [256]
    float* rsq    = (float*)(smem + 60928);        // [256]

    const int tid   = threadIdx.x;
    const int lane  = tid & 31;
    const int wid   = tid >> 5;
    const int lrow  = lane & 15;
    const int hi    = lane >> 4;
    const int mt    = wid >> 1;        // M-tile 0..3 (16 out channels each)
    const int halfw = wid & 1;         // which 4 N-tiles

    // --- stage weights (once) ---
    for (int i = tid; i < 8192; i += 256) {
        int c = i >> 7, k = i & 127;
        woL[c * ROWH + k] = (f16)wo[i];
    }
    if (tid < C_) {
        float run = 0.f;
        P_s[tid * 24] = 0.f;
#pragma unroll
        for (int k = 0; k < 23; ++k) { run += wc2[tid * 23 + k]; P_s[tid * 24 + k + 1] = run; }
        bc2_s[tid] = bc2[tid];
        bo_s[tid]  = bo[tid];
    }
    __syncthreads();

    // --- A fragments (wo) for this wave's M-tile, kept in VGPRs for whole WG ---
    v16h aW[4];
#pragma unroll
    for (int kf = 0; kf < 4; ++kf) {
        const f16* p = woL + (16 * mt + lrow) * ROWH + 32 * kf + 8 * hi;
        v8h lo = *(const v8h*)p;
        v8h hs = *(const v8h*)(p + 16);
#pragma unroll
        for (int e = 0; e < 8; ++e) { aW[kf][e] = lo[e]; aW[kf][8 + e] = hs[e]; }
    }

    // --- per-lane output bias (8 fixed channels), hoisted to registers ---
    v4f bo_lo = *(const v4f*)(bo_s + 16 * mt + 8 * hi);
    v4f bo_hi = *(const v4f*)(bo_s + 16 * mt + 8 * hi + 4);
    float bo_r[8];
#pragma unroll
    for (int e = 0; e < 4; ++e) { bo_r[e] = bo_lo[e]; bo_r[4 + e] = bo_hi[e]; }

    const int pb0 = blockIdx.x * (ITERS_D * 128);
    const int b   = pb0 >> 17;
    const int r00 = pb0 & (FT_ - 1);
    const float* xb = x     + (size_t)b * C_ * FT_;
    float*       ob = o_out + (size_t)b * C_ * FT_;

    float accS = 0.f, accQ = 0.f;      // running BN partials (threads < 64)

    for (int it = 0; it < ITERS_D; ++it) {
        const int r0    = r00 + it * 128;
        const int pbase = pb0 + it * 128;

        // phase 1: per-pixel att_map value + threshold count j
        if (tid < 128) {
            float a = att_map[pbase + tid];
            int j = 0;
#pragma unroll
            for (int k = 0; k < 23; ++k) {
                float thr = (float)((k + 1) * (k + 2)) * (1.0f / 600.0f);
                j += (a > thr) ? 1 : 0;
            }
            a_s[tid] = a; j_s[tid] = j;
        }
        __syncthreads();

        // phase 2: build B matrix in f16: rows 0..63 = x, rows 64..127 = att*x
        for (int i4 = tid; i4 < 2048; i4 += 256) {
            int c = i4 >> 5, t0 = (i4 & 31) * 4;
            const float* xp = xb + (size_t)c * FT_ + r0 + t0;
            v4f xv = *(const v4f*)xp;
            if (it < ITERS_D - 1) __builtin_prefetch(xp + 128, 0, 1);
            v4h hx, ha;
#pragma unroll
            for (int uu = 0; uu < 4; ++uu) {
                int t = t0 + uu;
                float attc = a_s[t] * P_s[c * 24 + j_s[t]] + bc2_s[c];
                hx[uu] = (f16)xv[uu];
                ha[uu] = (f16)(attc * xv[uu]);
            }
            *(v4h*)(bmat + c * ROWH + t0)          = hx;
            *(v4h*)(bmat + (64 + c) * ROWH + t0)   = ha;
        }
        __syncthreads();

        // phase 3: WMMA: each wave does its 4 N-tiles x 4 K-frags for its M-tile
        v8f acc[4];
#pragma unroll
        for (int nt = 0; nt < 4; ++nt) {
            int gnt = 4 * halfw + nt;
            v8f c8 = {0.f, 0.f, 0.f, 0.f, 0.f, 0.f, 0.f, 0.f};
#pragma unroll
            for (int kf = 0; kf < 4; ++kf) {
                const f16* bp = bmat + (32 * kf + 8 * hi) * ROWH + gnt * 16 + lrow;
                v16h bW;
#pragma unroll
                for (int e = 0; e < 8; ++e) { bW[e] = bp[e * ROWH]; bW[8 + e] = bp[(16 + e) * ROWH]; }
                c8 = __builtin_amdgcn_wmma_f32_16x16x32_f16(
                        false, aW[kf], false, bW, (short)0, c8, false, false);
            }
            acc[nt] = c8;
        }
        __syncthreads();   // all bmat reads done before o_tile (alias) writes

        // phase 4: spill D tiles (+bias, from registers) into o_tile
#pragma unroll
        for (int nt = 0; nt < 4; ++nt) {
            int gnt = 4 * halfw + nt;
#pragma unroll
            for (int r = 0; r < 8; ++r) {
                int ch = 16 * mt + 8 * hi + r;
                o_tile[ch * 128 + gnt * 16 + lrow] = acc[nt][r] + bo_r[r];
            }
        }
        __syncthreads();

        // phase 5: BN partial sums (deterministic fixed-order) + write o to global
        {
            int c = tid >> 2, q = tid & 3;
            float s = 0.f, ss = 0.f;
            const float* op = o_tile + c * 128 + q * 32;
#pragma unroll
            for (int i4 = 0; i4 < 8; ++i4) {
                v4f v = *(const v4f*)(op + i4 * 4);
#pragma unroll
                for (int e = 0; e < 4; ++e) { s += v[e]; ss += v[e] * v[e]; }
            }
            red[tid] = s; rsq[tid] = ss;
        }
        for (int i4 = tid; i4 < 2048; i4 += 256) {
            int c = i4 >> 5, t0 = (i4 & 31) * 4;
            *(v4f*)(ob + (size_t)c * FT_ + r0 + t0) = *(const v4f*)(o_tile + c * 128 + t0);
        }
        __syncthreads();
        if (tid < C_) {
            accS += red[4 * tid] + red[4 * tid + 1] + red[4 * tid + 2] + red[4 * tid + 3];
            accQ += rsq[4 * tid] + rsq[4 * tid + 1] + rsq[4 * tid + 2] + rsq[4 * tid + 3];
        }
        __syncthreads();
    }

    if (tid < C_) {
        partial[tid * NWG_D + blockIdx.x]        = accS;
        partial[(C_ + tid) * NWG_D + blockIdx.x] = accQ;
    }
}

// ---------------- Kernel 5a: reduce per-WG partials -> S[128] ------------------
__global__ __launch_bounds__(256) void k_reduce(const float* __restrict__ partial,
                                                float* __restrict__ S) {
    __shared__ float sd[256];
    const float* p = partial + blockIdx.x * NWG_D;
    int tid = threadIdx.x;
    float s = p[tid] + p[tid + 256] + p[tid + 512] + p[tid + 768];
    sd[tid] = s;
    __syncthreads();
    for (int off = 128; off >= 1; off >>= 1) {
        if (tid < off) sd[tid] += sd[tid + off];
        __syncthreads();
    }
    if (tid == 0) S[blockIdx.x] = sd[0];
}

// ---------------- Kernel 5b: finalize BN scale/shift ---------------------------
__global__ void k_finalize(const float* __restrict__ S,
                           const float* __restrict__ gamma,
                           const float* __restrict__ beta,
                           float* __restrict__ stats) {
    int c = threadIdx.x;           // 64 threads
    float mu  = S[c] * (1.0f / NP_);
    float var = S[C_ + c] * (1.0f / NP_) - mu * mu;
    float rs  = rsqrtf(var + 1e-5f);
    float sc  = rs * gamma[c];
    stats[c]      = sc;
    stats[C_ + c] = beta[c] - mu * sc;
}

// ---------------- Kernel 6: in-place BN + ReLU on o ----------------------------
__global__ __launch_bounds__(256) void k_bn_relu(float* __restrict__ o,
                                                 const float* __restrict__ stats) {
    const long total4 = (long)NP_ * C_ / 4;   // 16,777,216 float4s
    for (long i4 = (long)blockIdx.x * 256 + threadIdx.x; i4 < total4;
         i4 += (long)gridDim.x * 256) {
        int c = (int)((i4 >> 15) & 63);
        float sc = stats[c], sh = stats[C_ + c];
        v4f v = *(const v4f*)(o + i4 * 4);
        v4f r;
#pragma unroll
        for (int e = 0; e < 4; ++e) r[e] = fmaxf(v[e] * sc + sh, 0.f);
        *(v4f*)(o + i4 * 4) = r;
    }
}

extern "C" void kernel_launch(void* const* d_in, const int* in_sizes, int n_in,
                              void* d_out, int out_size, void* d_ws, size_t ws_size,
                              hipStream_t stream) {
    const float* x     = (const float*)d_in[0];
    const float* w1    = (const float*)d_in[1];
    const float* b1    = (const float*)d_in[2];
    const float* w2    = (const float*)d_in[3];
    const float* b2    = (const float*)d_in[4];
    const float* wc    = (const float*)d_in[5];
    const float* bc    = (const float*)d_in[6];
    const float* wc2   = (const float*)d_in[7];
    const float* bc2   = (const float*)d_in[8];
    const float* wo    = (const float*)d_in[9];
    const float* bo    = (const float*)d_in[10];
    const float* gamma = (const float*)d_in[11];
    const float* beta  = (const float*)d_in[12];
    float* out = (float*)d_out;

    float* ws      = (float*)d_ws;
    float* mF      = ws;                 // 262144
    float* mT      = ws + 262144;        // 131072
    float* zt      = ws + 393216;        // 262144
    float* u       = ws + 655360;        // 131072
    float* att     = ws + 786432;        // 1048576
    float* partial = ws + 1835008;       // 131072
    float* S       = ws + 1966080;       // 128
    float* stats   = ws + 1966208;       // 128

    k_row_mean <<<16384, 256, 0, stream>>>(x, mT);
    k_col_mean <<<1024,  256, 0, stream>>>(x, mF);
    k_mlp      <<<24,    256, 0, stream>>>(mF, mT, w1, b1, w2, b2, wc, zt, u);
    k_attmap   <<<1024,  256, 0, stream>>>(u, zt, bc, att);
    k_dtfa_main<<<NWG_D, 256, 0, stream>>>(x, att, wc2, bc2, wo, bo, out, partial);
    k_reduce   <<<128,   256, 0, stream>>>(partial, S);
    k_finalize <<<1,     64,  0, stream>>>(S, gamma, beta, stats);
    k_bn_relu  <<<4096,  256, 0, stream>>>(out, stats);
}
